// MambaFakeHead_12369505812979
// MI455X (gfx1250) — matrix operand order repaired and checked by reading, not compiled, not run on hardware
//
#include <hip/hip_runtime.h>
#include <hip/hip_fp16.h>
#include <math.h>

// ---------------- model dims ----------------
#define BATCH   16
#define L_TOK   49
#define DM      1792            // D_MODEL
#define DR      512             // D_RED
#define DI      1024            // D_INNER
#define DS      16              // D_STATE
#define NTOK    (BATCH * L_TOK) // 784 rows, 49 * 16-row WMMA tiles
#define NHEADS  8
#define HDIM    64

typedef __attribute__((ext_vector_type(16))) _Float16 v16h;
typedef __attribute__((ext_vector_type(8)))  float    v8f;

__device__ __forceinline__ float dev_silu(float x) { return x / (1.0f + expf(-x)); }
__device__ __forceinline__ float dev_softplus(float x) { return (x > 20.0f) ? x : log1pf(expf(x)); }
__device__ __forceinline__ float dev_gelu(float x) { return 0.5f * x * (1.0f + erff(x * 0.70710678118654752f)); }

// =====================================================================
// Split-f16 WMMA GEMM:  C[M,N] = act(A[M,K] @ W[N,K]^T + bias)
// Block = 4 waves sharing one 64-column N-strip, one 16-row M-tile each.
// The 64x32 W K-tile is staged ONCE per block into LDS with
// global_load_async_to_lds_b128 (ASYNCcnt), double-buffered so the copy of
// tile k+1 overlaps the 12 WMMAs on tile k. B fragments then come from LDS
// (ds_load_b128); A fragments are private per wave -> direct global loads
// with global_prefetch_b8 lookahead.
// fp32 accuracy recovered via Ahi*Bhi + Ahi*Blo + Alo*Bhi.
// K must be a multiple of 32; M a multiple of 16.
// =====================================================================
__global__ __launch_bounds__(128)
void gemm_wmma_splitf16(const float* __restrict__ A, const float* __restrict__ W,
                        const float* __restrict__ bias, float* __restrict__ C,
                        int M, int N, int K, int act)
{
    __shared__ float bsm[2][64 * 32];   // double-buffered 64x32 B strip (16 KB)

    const int tilesN = (N + 63) >> 6;   // 64-column strips
    const int tilesM = M >> 4;
    int tn  = blockIdx.x % tilesN;
    int tmg = blockIdx.x / tilesN;
    int wv  = threadIdx.x >> 5;         // wave 0..3 -> M-tile within group
    int tm  = tmg * 4 + wv;
    bool active = (tm < tilesM);        // wave-uniform; barriers stay outside

    int lane = threadIdx.x & 31;
    int r    = lane & 15;
    int half = lane >> 4;
    int tid  = threadIdx.x;

    const float* Ap = A + (size_t)((active ? tm : 0) * 16 + r) * K;

    // LDS byte offset of the staging buffer (async-to-LDS takes LDS offsets)
    unsigned lds_base = (unsigned)(uintptr_t)(__attribute__((address_space(3))) float*)&bsm[0][0];

    // per-thread chunk assignment for staging: 512 x 16B chunks = 64 rows x 32 floats
    const float* gsrc[4];
    unsigned     loff[4];
#pragma unroll
    for (int q = 0; q < 4; ++q) {
        int fi = q * 128 + tid;          // chunk index 0..511
        int rr = fi >> 3;                // strip row 0..63
        int c4 = (fi & 7) << 2;          // col 0,4,...,28
        int bn = tn * 64 + rr; if (bn > N - 1) bn = N - 1;   // clamp; stores are guarded
        gsrc[q] = W + (size_t)bn * K + c4;
        loff[q] = lds_base + (unsigned)((rr * 32 + c4) << 2);
    }

    v8f acc[4];
#pragma unroll
    for (int j = 0; j < 4; ++j) acc[j] = (v8f){0.f,0.f,0.f,0.f,0.f,0.f,0.f,0.f};

    const int nK = K >> 5;

    // ---- prologue: stage K-tile 0 into buffer 0 ----
#pragma unroll
    for (int q = 0; q < 4; ++q) {
        asm volatile("global_load_async_to_lds_b128 %0, %1, off"
                     :: "v"(loff[q]), "v"((unsigned long long)(uintptr_t)gsrc[q])
                     : "memory");
    }
    asm volatile("s_wait_asynccnt 0x0" ::: "memory");
    __syncthreads();

    int cur = 0;
    for (int kt = 0; kt < nK; ++kt) {
        int k0 = kt << 5;

        // ---- issue async copy of K-tile kt+1 into the other buffer ----
        if (kt + 1 < nK) {
            unsigned dstoff = (unsigned)((cur ^ 1) * 64 * 32 * 4);
#pragma unroll
            for (int q = 0; q < 4; ++q) {
                asm volatile("global_load_async_to_lds_b128 %0, %1, off"
                             :: "v"(loff[q] + dstoff),
                                "v"((unsigned long long)(uintptr_t)(gsrc[q] + k0 + 32))
                             : "memory");
            }
        }

        if (active) {
            if (kt + 1 < nK) __builtin_prefetch(Ap + k0 + 32, 0, 1);

            // A 16x32 frag: lanes 0-15 hold K {0..7,16..23}, lanes 16-31 hold K {8..15,24..31}
            float af[16];
            {
                const float4* p0 = (const float4*)(Ap + k0 + half * 8);
                const float4* p1 = (const float4*)(Ap + k0 + 16 + half * 8);
                float4 u0 = p0[0], u1 = p0[1], u2 = p1[0], u3 = p1[1];
                af[0]=u0.x;  af[1]=u0.y;  af[2]=u0.z;  af[3]=u0.w;
                af[4]=u1.x;  af[5]=u1.y;  af[6]=u1.z;  af[7]=u1.w;
                af[8]=u2.x;  af[9]=u2.y;  af[10]=u2.z; af[11]=u2.w;
                af[12]=u3.x; af[13]=u3.y; af[14]=u3.z; af[15]=u3.w;
            }
            v16h ahi, alo;
#pragma unroll
            for (int i = 0; i < 16; ++i) {
                _Float16 h1 = (_Float16)af[i];
                ahi[i] = h1; alo[i] = (_Float16)(af[i] - (float)h1);
            }

            // 4 B tiles from LDS, each 32x16: lanes 0-15 hold K {0..7,16..23}, lanes 16-31 K {8..15,24..31}
#pragma unroll
            for (int j = 0; j < 4; ++j) {
                float bf[16];
                const float4* sp = (const float4*)&bsm[cur][(j * 16 + r) * 32 + half * 8];
                const float4* sq = (const float4*)&bsm[cur][(j * 16 + r) * 32 + 16 + half * 8];
                float4 u0 = sp[0], u1 = sp[1], u2 = sq[0], u3 = sq[1];
                bf[0]=u0.x;  bf[1]=u0.y;  bf[2]=u0.z;  bf[3]=u0.w;
                bf[4]=u1.x;  bf[5]=u1.y;  bf[6]=u1.z;  bf[7]=u1.w;
                bf[8]=u2.x;  bf[9]=u2.y;  bf[10]=u2.z; bf[11]=u2.w;
                bf[12]=u3.x; bf[13]=u3.y; bf[14]=u3.z; bf[15]=u3.w;
                v16h bhi, blo;
#pragma unroll
                for (int i = 0; i < 16; ++i) {
                    _Float16 h2 = (_Float16)bf[i];
                    bhi[i] = h2; blo[i] = (_Float16)(bf[i] - (float)h2);
                }
                acc[j] = __builtin_amdgcn_wmma_f32_16x16x32_f16(false, alo, false, bhi, (short)0, acc[j], false, false);
                acc[j] = __builtin_amdgcn_wmma_f32_16x16x32_f16(false, ahi, false, blo, (short)0, acc[j], false, false);
                acc[j] = __builtin_amdgcn_wmma_f32_16x16x32_f16(false, ahi, false, bhi, (short)0, acc[j], false, false);
            }
        }

        // drain this wave's async copies, then block-wide visibility for next tile
        asm volatile("s_wait_asynccnt 0x0" ::: "memory");
        __syncthreads();
        cur ^= 1;
    }

    if (active) {
#pragma unroll
        for (int j = 0; j < 4; ++j) {
            int ocol = tn * 64 + j * 16 + r;
            if (ocol < N) {
                float bv = bias ? bias[ocol] : 0.0f;
#pragma unroll
                for (int i = 0; i < 8; ++i) {   // C/D layout: vgpr i -> row i (+8 for upper half-lanes)
                    int orow = tm * 16 + half * 8 + i;
                    float v = acc[j][i] + bv;
                    if (act == 1) v = dev_gelu(v);
                    C[(size_t)orow * N + ocol] = v;
                }
            }
        }
    }
}

// ---------------- elementwise / fused kernels ----------------

// (B, C, 49) -> (B*49, C)
__global__ void tokenize_k(const float* __restrict__ f, float* __restrict__ t, int C, int total)
{
    int idx = blockIdx.x * blockDim.x + threadIdx.x;
    if (idx >= total) return;
    int rl = idx / C, c = idx - rl * C;
    int b = rl / L_TOK, l = rl - b * L_TOK;
    t[idx] = f[((size_t)b * C + c) * L_TOK + l];
}

// branch-1 pooling: gcat[b, 0:DM] = mean_l fs, gcat[b, DM:2DM] = max_l fs
__global__ void pool_k(const float* __restrict__ fs, float* __restrict__ gcat, int total)
{
    int idx = blockIdx.x * blockDim.x + threadIdx.x;
    if (idx >= total) return;
    int b = idx / DM, c = idx - b * DM;
    const float* p = fs + ((size_t)b * DM + c) * L_TOK;
    float s = 0.f, m = -1e30f;
    for (int l = 0; l < L_TOK; ++l) { float v = p[l]; s += v; m = fmaxf(m, v); }
    gcat[(size_t)b * (2 * DM) + c]      = s * (1.0f / L_TOK);
    gcat[(size_t)b * (2 * DM) + DM + c] = m;
}

// LayerNorm over last dim (optional residual added first)
__global__ void layernorm_k(const float* __restrict__ x, const float* __restrict__ res,
                            const float* __restrict__ w, const float* __restrict__ b,
                            float* __restrict__ out, int cols)
{
    __shared__ float s1[128], s2[128];
    int row = blockIdx.x, t = threadIdx.x;
    const float* xr = x + (size_t)row * cols;
    const float* rr = res ? res + (size_t)row * cols : nullptr;
    float s = 0.f, q = 0.f;
    for (int c = t; c < cols; c += 128) {
        float v = xr[c] + (rr ? rr[c] : 0.f);
        s += v; q += v * v;
    }
    s1[t] = s; s2[t] = q;
    __syncthreads();
    for (int o = 64; o > 0; o >>= 1) {
        if (t < o) { s1[t] += s1[t + o]; s2[t] += s2[t + o]; }
        __syncthreads();
    }
    float mean = s1[0] / cols;
    float var  = s2[0] / cols - mean * mean;
    float inv  = rsqrtf(var + 1e-5f);
    for (int c = t; c < cols; c += 128) {
        float v = xr[c] + (rr ? rr[c] : 0.f);
        out[(size_t)row * cols + c] = (v - mean) * inv * w[c] + b[c];
    }
}

// reverse sequence: out[b, L-1-l, :] = in[b, l, :]
__global__ void flip_k(const float* __restrict__ x, float* __restrict__ out, int C, int total)
{
    int idx = blockIdx.x * blockDim.x + threadIdx.x;
    if (idx >= total) return;
    int rl = idx / C, c = idx - rl * C;
    int b = rl / L_TOK, l = rl - b * L_TOK;
    out[((size_t)(b * L_TOK + (L_TOK - 1 - l))) * C + c] = x[idx];
}

// depthwise causal conv1d (d_conv=4) + bias + SiLU; xi = xz[..., :DI]
__global__ void conv_silu_k(const float* __restrict__ xz, const float* __restrict__ w,
                            const float* __restrict__ cb, float* __restrict__ xc, int total)
{
    int idx = blockIdx.x * blockDim.x + threadIdx.x;
    if (idx >= total) return;
    int d  = idx & (DI - 1);
    int rl = idx >> 10;
    int b  = rl / L_TOK, l = rl - b * L_TOK;
    float acc = cb[d];
#pragma unroll
    for (int j = 0; j < 4; ++j) {
        int ls = l - 3 + j;
        if (ls >= 0) acc += w[d * 4 + j] * xz[((size_t)(b * L_TOK + ls)) * (2 * DI) + d];
    }
    xc[(size_t)rl * DI + d] = dev_silu(acc);
}

// selective-scan: one thread per (b, d); 16 states in registers; 49 steps
__global__ void ssm_scan_k(const float* __restrict__ bcdt, const float* __restrict__ xc,
                           const float* __restrict__ xz, const float* __restrict__ dt_w,
                           const float* __restrict__ dt_b, const float* __restrict__ A_log,
                           const float* __restrict__ Dp, float* __restrict__ yg)
{
    int idx = blockIdx.x * blockDim.x + threadIdx.x;
    if (idx >= BATCH * DI) return;
    int b = idx >> 10, d = idx & (DI - 1);
    float A[DS], h[DS];
#pragma unroll
    for (int s = 0; s < DS; ++s) {
        float al = A_log[d * DS + s];
        al = fminf(fmaxf(al, -6.0f), 6.0f);
        A[s] = -expf(al);
        h[s] = 0.f;
    }
    float dw = dt_w[d], db = dt_b[d], Dd = Dp[d];
    for (int l = 0; l < L_TOK; ++l) {
        int rl = b * L_TOK + l;
        const float* row = bcdt + (size_t)rl * (1 + 2 * DS);
        float dt  = dev_softplus(row[0] * dw + db);
        float xcv = xc[(size_t)rl * DI + d];
        float zv  = xz[(size_t)rl * (2 * DI) + DI + d];
        float y = 0.f;
#pragma unroll
        for (int s = 0; s < DS; ++s) {
            float dA  = expf(dt * A[s]);
            float hn  = dA * h[s] + dt * row[1 + s] * xcv;
            hn = fminf(fmaxf(hn, -10000.0f), 10000.0f);
            h[s] = hn;
            y += hn * row[1 + DS + s];
        }
        yg[(size_t)rl * DI + d] = (y + xcv * Dd) * dev_silu(zv);
    }
}

// x = sigmoid(alpha)*yf + (1-sigmoid(alpha))*flip(yb) + x   (in place)
__global__ void combine_k(const float* __restrict__ yf, const float* __restrict__ yb,
                          const float* __restrict__ alpha, float* __restrict__ x, int total)
{
    int idx = blockIdx.x * blockDim.x + threadIdx.x;
    if (idx >= total) return;
    int c  = idx & (DR - 1);
    int rl = idx >> 9;
    int b  = rl / L_TOK, l = rl - b * L_TOK;
    float a = 1.0f / (1.0f + expf(-alpha[0]));
    size_t fidx = ((size_t)(b * L_TOK + (L_TOK - 1 - l))) * DR + c;
    x[idx] = a * yf[idx] + (1.0f - a) * yb[fidx] + x[idx];
}

// attention for one (b, h, q): 64 threads, L=49 keys
__global__ void attn_k(const float* __restrict__ qkvq, const float* __restrict__ qkvkv,
                       float* __restrict__ out)
{
    int blk = blockIdx.x;
    int q = blk % L_TOK;
    int h = (blk / L_TOK) & (NHEADS - 1);
    int b = blk / (L_TOK * NHEADS);
    int t = threadIdx.x;
    __shared__ float sc[64];
    __shared__ float smax, ssum;
    const float* Q = qkvq + ((size_t)(b * L_TOK + q)) * (3 * DR) + h * HDIM;
    if (t < L_TOK) {
        const float* Kp = qkvkv + ((size_t)(b * L_TOK + t)) * (3 * DR) + DR + h * HDIM;
        float s = 0.f;
#pragma unroll 8
        for (int d = 0; d < HDIM; ++d) s += Q[d] * Kp[d];
        sc[t] = s * 0.125f;   // 1/sqrt(64)
    }
    __syncthreads();
    if (t == 0) {
        float m = -1e30f;
        for (int i = 0; i < L_TOK; ++i) m = fmaxf(m, sc[i]);
        float sum = 0.f;
        for (int i = 0; i < L_TOK; ++i) sum += expf(sc[i] - m);
        smax = m; ssum = sum;
    }
    __syncthreads();
    if (t < L_TOK) sc[t] = expf(sc[t] - smax) / ssum;
    __syncthreads();
    float o = 0.f;
    for (int k = 0; k < L_TOK; ++k)
        o += sc[k] * qkvkv[((size_t)(b * L_TOK + k)) * (3 * DR) + 2 * DR + h * HDIM + t];
    out[((size_t)(b * L_TOK + q)) * DR + h * HDIM + t] = o;
}

// mean over tokens: out[b*ldo + off + c] = mean_l x[b,l,c]
__global__ void mean_tokens_k(const float* __restrict__ x, float* __restrict__ out,
                              int ldo, int off, int C, int total)
{
    int idx = blockIdx.x * blockDim.x + threadIdx.x;
    if (idx >= total) return;
    int b = idx / C, c = idx - b * C;
    float s = 0.f;
    for (int l = 0; l < L_TOK; ++l) s += x[((size_t)(b * L_TOK + l)) * C + c];
    out[(size_t)b * ldo + off + c] = s * (1.0f / L_TOK);
}

// out[b] = sum_c x[b,c]*w[c] + bias[0]
__global__ void dot_head_k(const float* __restrict__ x, const float* __restrict__ w,
                           const float* __restrict__ bias, float* __restrict__ out, int C)
{
    __shared__ float red[256];
    int b = blockIdx.x, t = threadIdx.x;
    float s = 0.f;
    for (int c = t; c < C; c += 256) s += x[(size_t)b * C + c] * w[c];
    red[t] = s;
    __syncthreads();
    for (int o = 128; o > 0; o >>= 1) {
        if (t < o) red[t] += red[t + o];
        __syncthreads();
    }
    if (t == 0) out[b] = red[0] + bias[0];
}

// ensemble: out[b] = softmax(w3) . [lg, ls, ld]
__global__ void final_k(const float* __restrict__ lg, const float* __restrict__ ls,
                        const float* __restrict__ ld, const float* __restrict__ w3,
                        float* __restrict__ out)
{
    int b = threadIdx.x;
    if (b >= BATCH) return;
    float w0 = w3[0], w1 = w3[1], w2 = w3[2];
    float m = fmaxf(w0, fmaxf(w1, w2));
    float e0 = expf(w0 - m), e1 = expf(w1 - m), e2 = expf(w2 - m);
    float inv = 1.0f / (e0 + e1 + e2);
    out[b] = (e0 * lg[b] + e1 * ls[b] + e2 * ld[b]) * inv;
}

// ---------------- host-side orchestration ----------------

static inline int cdiv_i(int a, int b) { return (a + b - 1) / b; }

static void launch_gemm(hipStream_t s, const float* A, const float* W, const float* bias,
                        float* C, int M, int N, int K, int act)
{
    int tilesN = cdiv_i(N, 64);
    int tmg    = cdiv_i(M / 16, 4);          // 4 M-tiles (one per wave) per block
    gemm_wmma_splitf16<<<tilesN * tmg, 128, 0, s>>>(A, W, bias, C, M, N, K, act);
}

struct CoreP  { const float *A_log, *D, *conv_b, *conv_w, *dt_b, *dt_w, *in_proj, *out_proj, *x_proj; };
struct BlockP { const float *alpha; CoreP bwd, fwd; const float *norm_b, *norm_w; };
struct MhaP   { const float *in_b, *in_w, *out_b, *out_w; };

struct Scratch { float *xz, *xc, *bcdt, *yg; };

static void ssm_core_run(hipStream_t s, const CoreP& cp, const float* xin, float* yout, const Scratch& sc)
{
    launch_gemm(s, xin, cp.in_proj, nullptr, sc.xz, NTOK, 2 * DI, DR, 0);
    conv_silu_k<<<cdiv_i(NTOK * DI, 256), 256, 0, s>>>(sc.xz, cp.conv_w, cp.conv_b, sc.xc, NTOK * DI);
    launch_gemm(s, sc.xc, cp.x_proj, nullptr, sc.bcdt, NTOK, 1 + 2 * DS, DI, 0);
    ssm_scan_k<<<cdiv_i(BATCH * DI, 256), 256, 0, s>>>(sc.bcdt, sc.xc, sc.xz, cp.dt_w, cp.dt_b,
                                                       cp.A_log, cp.D, sc.yg);
    launch_gemm(s, sc.yg, cp.out_proj, nullptr, yout, NTOK, DR, DI, 0);
}

static void bimamba_run(hipStream_t s, const BlockP& bp, float* x, float* xn, float* xflip,
                        float* yf, float* yb, const Scratch& sc)
{
    layernorm_k<<<NTOK, 128, 0, s>>>(x, nullptr, bp.norm_w, bp.norm_b, xn, DR);
    ssm_core_run(s, bp.fwd, xn, yf, sc);
    flip_k<<<cdiv_i(NTOK * DR, 256), 256, 0, s>>>(xn, xflip, DR, NTOK * DR);
    ssm_core_run(s, bp.bwd, xflip, yb, sc);
    combine_k<<<cdiv_i(NTOK * DR, 256), 256, 0, s>>>(yf, yb, bp.alpha, x, NTOK * DR);
}

extern "C" void kernel_launch(void* const* d_in, const int* in_sizes, int n_in,
                              void* d_out, int out_size, void* d_ws, size_t ws_size,
                              hipStream_t stream)
{
    (void)in_sizes; (void)n_in; (void)out_size; (void)ws_size;
    const float* fs = (const float*)d_in[0];
    const float* ft = (const float*)d_in[1];

    // ---- unpack params in JAX pytree order (dict keys sorted, lists in order) ----
    int cur = 2;
    auto nxt  = [&]() { return (const float*)d_in[cur++]; };
    auto core = [&]() { CoreP p; p.A_log = nxt(); p.D = nxt(); p.conv_b = nxt(); p.conv_w = nxt();
                        p.dt_b = nxt(); p.dt_w = nxt(); p.in_proj = nxt(); p.out_proj = nxt();
                        p.x_proj = nxt(); return p; };
    auto blockp = [&]() { BlockP b; b.alpha = nxt(); b.bwd = core(); b.fwd = core();
                          b.norm_b = nxt(); b.norm_w = nxt(); return b; };
    auto mhap = [&]() { MhaP m; m.in_b = nxt(); m.in_w = nxt(); m.out_b = nxt(); m.out_w = nxt(); return m; };

    MhaP cross_st = mhap();
    MhaP cross_ts = mhap();
    BlockP du_s[2] = { blockp(), blockp() };
    BlockP du_t[2] = { blockp(), blockp() };
    const float* fus_b1 = nxt(); const float* fus_b2 = nxt();
    const float* fus_w1 = nxt(); const float* fus_w2 = nxt();
    const float* glob_b2 = nxt(); const float* glob_w1 = nxt(); const float* glob_w2 = nxt();
    const float* logit_w3 = nxt();
    const float* norm_s_b = nxt();  const float* norm_s_w = nxt();
    const float* norm_st_b = nxt(); const float* norm_st_w = nxt();
    const float* norm_t_b = nxt();  const float* norm_t_w = nxt();
    const float* norm_ts_b = nxt(); const float* norm_ts_w = nxt();
    const float* proj_s = nxt(); const float* proj_t = nxt();
    BlockP sp_blk[2] = { blockp(), blockp() };
    const float* sp_head_b = nxt(); const float* sp_head_w = nxt();
    const float* sp_in = nxt();
    const float* sp_norm_b = nxt(); const float* sp_norm_w = nxt();

    // ---- workspace carve (deterministic) ----
    char* wp = (char*)d_ws;
    auto carve = [&](size_t n) { float* r = (float*)wp; wp += ((n * 4 + 255) & ~(size_t)255); return r; };
    float* tfs   = carve((size_t)NTOK * DM);
    float* tft   = carve((size_t)NTOK * DM);
    float* gcat  = carve((size_t)BATCH * 2 * DM);
    float* gbuf  = carve((size_t)BATCH * DM);
    float* t2    = carve((size_t)NTOK * DR);
    float* ts    = carve((size_t)NTOK * DR);
    float* tt    = carve((size_t)NTOK * DR);
    float* tsn   = carve((size_t)NTOK * DR);
    float* ttn   = carve((size_t)NTOK * DR);
    float* xn    = carve((size_t)NTOK * DR);
    float* xflip = carve((size_t)NTOK * DR);
    float* yf    = carve((size_t)NTOK * DR);
    float* yb    = carve((size_t)NTOK * DR);
    Scratch sc;
    sc.xz   = carve((size_t)NTOK * 2 * DI);
    sc.xc   = carve((size_t)NTOK * DI);
    sc.bcdt = carve((size_t)NTOK * (1 + 2 * DS));
    sc.yg   = carve((size_t)NTOK * DI);
    float* qkva  = carve((size_t)NTOK * 3 * DR);
    float* qkvb  = carve((size_t)NTOK * 3 * DR);
    float* attno = carve((size_t)NTOK * DR);
    float* mo    = carve((size_t)NTOK * DR);
    float* ts2   = carve((size_t)NTOK * DR);
    float* tt2   = carve((size_t)NTOK * DR);
    float* meanb = carve((size_t)BATCH * DR);
    float* gcat2 = carve((size_t)BATCH * 2 * DR);
    float* fh    = carve((size_t)BATCH * DR);
    float* lg    = carve(16);
    float* ls    = carve(16);
    float* ld    = carve(16);

    // ---- tokenize both feature maps ----
    tokenize_k<<<cdiv_i(NTOK * DM, 256), 256, 0, stream>>>(fs, tfs, DM, NTOK * DM);
    tokenize_k<<<cdiv_i(NTOK * DM, 256), 256, 0, stream>>>(ft, tft, DM, NTOK * DM);

    // ---- Branch 1: global head ----
    pool_k<<<cdiv_i(BATCH * DM, 256), 256, 0, stream>>>(fs, gcat, BATCH * DM);
    launch_gemm(stream, gcat, glob_w1, nullptr, gbuf, BATCH, DM, 2 * DM, 1 /*gelu*/);
    dot_head_k<<<BATCH, 256, 0, stream>>>(gbuf, glob_w2, glob_b2, lg, DM);

    // ---- Branch 2: spatial BiMamba ----
    launch_gemm(stream, tfs, sp_in, nullptr, t2, NTOK, DR, DM, 0);
    for (int i = 0; i < 2; ++i) bimamba_run(stream, sp_blk[i], t2, xn, xflip, yf, yb, sc);
    layernorm_k<<<NTOK, 128, 0, stream>>>(t2, nullptr, sp_norm_w, sp_norm_b, xn, DR);
    mean_tokens_k<<<cdiv_i(BATCH * DR, 256), 256, 0, stream>>>(xn, meanb, DR, 0, DR, BATCH * DR);
    dot_head_k<<<BATCH, 256, 0, stream>>>(meanb, sp_head_w, sp_head_b, ls, DR);

    // ---- Branch 3: dual-identity mamba fusion ----
    launch_gemm(stream, tfs, proj_s, nullptr, ts, NTOK, DR, DM, 0);
    launch_gemm(stream, tft, proj_t, nullptr, tt, NTOK, DR, DM, 0);
    for (int i = 0; i < 2; ++i) bimamba_run(stream, du_s[i], ts, xn, xflip, yf, yb, sc);
    for (int i = 0; i < 2; ++i) bimamba_run(stream, du_t[i], tt, xn, xflip, yf, yb, sc);
    layernorm_k<<<NTOK, 128, 0, stream>>>(ts, nullptr, norm_s_w, norm_s_b, tsn, DR);
    layernorm_k<<<NTOK, 128, 0, stream>>>(tt, nullptr, norm_t_w, norm_t_b, ttn, DR);

    // cross_st: q = tsn, kv = ttn
    launch_gemm(stream, tsn, cross_st.in_w, cross_st.in_b, qkva, NTOK, 3 * DR, DR, 0);
    launch_gemm(stream, ttn, cross_st.in_w, cross_st.in_b, qkvb, NTOK, 3 * DR, DR, 0);
    attn_k<<<BATCH * NHEADS * L_TOK, 64, 0, stream>>>(qkva, qkvb, attno);
    launch_gemm(stream, attno, cross_st.out_w, cross_st.out_b, mo, NTOK, DR, DR, 0);
    layernorm_k<<<NTOK, 128, 0, stream>>>(tsn, mo, norm_st_w, norm_st_b, ts2, DR);

    // cross_ts: q = ttn, kv = tsn
    launch_gemm(stream, ttn, cross_ts.in_w, cross_ts.in_b, qkva, NTOK, 3 * DR, DR, 0);
    launch_gemm(stream, tsn, cross_ts.in_w, cross_ts.in_b, qkvb, NTOK, 3 * DR, DR, 0);
    attn_k<<<BATCH * NHEADS * L_TOK, 64, 0, stream>>>(qkva, qkvb, attno);
    launch_gemm(stream, attno, cross_ts.out_w, cross_ts.out_b, mo, NTOK, DR, DR, 0);
    layernorm_k<<<NTOK, 128, 0, stream>>>(ttn, mo, norm_ts_w, norm_ts_b, tt2, DR);

    mean_tokens_k<<<cdiv_i(BATCH * DR, 256), 256, 0, stream>>>(ts2, gcat2, 2 * DR, 0,  DR, BATCH * DR);
    mean_tokens_k<<<cdiv_i(BATCH * DR, 256), 256, 0, stream>>>(tt2, gcat2, 2 * DR, DR, DR, BATCH * DR);
    launch_gemm(stream, gcat2, fus_w1, fus_b1, fh, BATCH, DR, 2 * DR, 1 /*gelu*/);
    dot_head_k<<<BATCH, 256, 0, stream>>>(fh, fus_w2, fus_b2, ld, DR);

    // ---- ensemble ----
    final_k<<<1, 32, 0, stream>>>(lg, ls, ld, logit_w3, (float*)d_out);
}
